// GPT_10222022165106
// MI455X (gfx1250) — matrix operand order repaired and checked by reading, not compiled, not run on hardware
//
#include <hip/hip_runtime.h>
#include <cstddef>
#include <cstdint>

// ---------------- problem constants (match reference) ----------------
constexpr int Bz  = 2;
constexpr int Tz  = 1024;
constexpr int Cz  = 768;
constexpr int Hz  = 12;
constexpr int Lz  = 4;
constexpr int Vz  = 50257;
constexpr int HDz = Cz / Hz;   // 64
constexpr int Mz  = Bz * Tz;   // 2048 rows of activations

// ---------------- types ----------------
typedef __bf16 bf16;
typedef __attribute__((ext_vector_type(8)))  bf16  bf16x8;
typedef __attribute__((ext_vector_type(16))) bf16  bf16x16;
typedef __attribute__((ext_vector_type(8)))  float f32x8;

#define SHUF16(lo, hi) __builtin_shufflevector((lo), (hi), 0,1,2,3,4,5,6,7,8,9,10,11,12,13,14,15)

static __device__ __forceinline__ f32x8 wmma_bf16(bf16x16 a, bf16x16 b, f32x8 c) {
  // D = A(16x32) * B(32x16) + C, fp32 accumulate — v_wmma_f32_16x16x32_bf16
  return __builtin_amdgcn_wmma_f32_16x16x32_bf16(false, a, false, b, (short)0, c, false, false);
}

static __device__ __forceinline__ float gelu_tanh(float x) {
  float u = 0.7978845608028654f * (x + 0.044715f * x * x * x);
  return 0.5f * x * (1.0f + tanhf(u));
}

// ---------------- CDNA5 TDM / wait support ----------------
#if defined(__has_builtin)
#  if __has_builtin(__builtin_amdgcn_tensor_load_to_lds)
#    define HAVE_TDM 1
#  endif
#  if __has_builtin(__builtin_amdgcn_s_wait_tensorcnt)
#    define HAVE_WTC 1
#  endif
#endif
#ifndef HAVE_TDM
#  define HAVE_TDM 0
#endif
#ifndef HAVE_WTC
#  define HAVE_WTC 0
#endif

static __device__ __forceinline__ void wait_tensor0() {
#if HAVE_WTC
  __builtin_amdgcn_s_wait_tensorcnt((short)0);
#else
  asm volatile("s_wait_tensorcnt 0" ::: "memory");
#endif
}

constexpr int BM = 64, BN = 64, BK = 32;
constexpr int LDT = BK + 8;   // padded LDS row in halves: 80B stride, 16B-aligned chunks

#if HAVE_TDM
typedef unsigned int u32x4 __attribute__((ext_vector_type(4)));
typedef int          i32x8 __attribute__((ext_vector_type(8)));
typedef int          i32x4 __attribute__((ext_vector_type(4)));

// 2D TDM tile load: 64 rows x 32 bf16 from row-major [.,K] source into LDS with
// 8-half padding after each 32-half row (matches LDT=40 layout).
static __device__ __forceinline__ void tdm_load_tile(const bf16* gsrc, bf16* ldst, int K) {
  const unsigned long long ga = (unsigned long long)(uintptr_t)gsrc;
  u32x4 g0;
  g0[0] = 1u;                                               // count=1, user descriptor
  g0[1] = (unsigned)(uintptr_t)ldst;                        // LDS byte address
  g0[2] = (unsigned)(ga & 0xFFFFFFFFull);                   // global addr [31:0]
  g0[3] = (unsigned)((ga >> 32) & 0x1FFFFFFull) | (2u << 30); // addr[56:32] | type=2
  i32x8 g1;
  // data_size=2B (code 1), pad_enable, pad_interval=16 DWORDs (code 3 => 32 halves),
  // pad_amount=4 DWORDs (code 3 => 8 halves)
  g1[0] = (int)((1u << 16) | (1u << 20) | (3u << 22) | (3u << 25));
  g1[1] = (int)(((unsigned)K & 0xFFFFu) << 16);             // tensor_dim0[15:0]
  g1[2] = (int)((((unsigned)K >> 16) & 0xFFFFu) | (0xFFFFu << 16)); // dim0[31:16] | dim1[15:0]
  g1[3] = (int)(0x7FFFu | ((unsigned)BK << 16));            // dim1[31:16] | tile_dim0=32
  g1[4] = (int)(unsigned)BN;                                // tile_dim1=64, tile_dim2=0
  g1[5] = (int)(unsigned)K;                                 // tensor_dim0_stride[31:0] (elements)
  g1[6] = 0;                                                // stride0[47:32] | stride1 lo
  g1[7] = 0;
  i32x4 z4 = {0, 0, 0, 0};
  i32x8 z8 = {0, 0, 0, 0, 0, 0, 0, 0};
  __builtin_amdgcn_tensor_load_to_lds(g0, g1, z4, z4, z8, 0);
}
#endif

// ---------------- embedding: x = wte[idx] + wpe ----------------
__global__ __launch_bounds__(256)
void embed_kernel(const int* __restrict__ idx, const float* __restrict__ wte,
                  const float* __restrict__ wpe, float* __restrict__ x) {
  const int bt  = blockIdx.x;
  const int t   = bt % Tz;
  const int tok = idx[bt];
  const float* e = wte + (size_t)tok * Cz;
  const float* p = wpe + (size_t)t * Cz;
  float* xr = x + (size_t)bt * Cz;
  for (int c = threadIdx.x; c < Cz; c += 256) xr[c] = e[c] + p[c];
}

// ---------------- fp32[K,N] -> bf16[N,K] weight convert+transpose ----------------
__global__ __launch_bounds__(256)
void wconv_kernel(const float* __restrict__ W, bf16* __restrict__ Wt, int K, int N) {
  const int gid = blockIdx.x * 256 + threadIdx.x;
  const int kg  = K / 8;
  if (gid >= N * kg) return;
  const int n  = gid / kg;
  const int k8 = (gid % kg) * 8;
  bf16x8 v;
  #pragma unroll
  for (int i = 0; i < 8; ++i) v[i] = (bf16)W[(size_t)(k8 + i) * N + n];
  *(bf16x8*)(Wt + (size_t)n * K + k8) = v;
}

// ---------------- layernorm (block per row, C=768) ----------------
__global__ __launch_bounds__(256)
void ln_kernel(const float* __restrict__ x, const float* __restrict__ g,
               const float* __restrict__ b, float* __restrict__ y, float eps) {
  const int row = blockIdx.x;
  const float* xr = x + (size_t)row * Cz;
  float* yr = y + (size_t)row * Cz;
  __shared__ float red0[8], red1[8];
  float s = 0.f, q = 0.f;
  for (int c = threadIdx.x; c < Cz; c += 256) { float v = xr[c]; s += v; q += v * v; }
  #pragma unroll
  for (int off = 16; off >= 1; off >>= 1) {
    s += __shfl_xor(s, off, 32);
    q += __shfl_xor(q, off, 32);
  }
  const int wv = threadIdx.x >> 5;
  if ((threadIdx.x & 31) == 0) { red0[wv] = s; red1[wv] = q; }
  __syncthreads();
  s = 0.f; q = 0.f;
  #pragma unroll
  for (int w = 0; w < 8; ++w) { s += red0[w]; q += red1[w]; }
  const float mu   = s * (1.0f / Cz);
  const float var  = q * (1.0f / Cz) - mu * mu;
  const float rstd = rsqrtf(var + eps);
  for (int c = threadIdx.x; c < Cz; c += 256)
    yr[c] = (xr[c] - mu) * rstd * g[c] + b[c];
}

// ---------------- WMMA GEMM: out = epi(A[M,K] @ W^T + bias [+resid]) ----------------
// Weight is [N,K]. BF16W=true: bf16 weights staged by TDM (double-buffered, one
// barrier per K-step). BF16W=false: fp32 weights (wte) converted on the fly.
// EPI: 0 = none, 1 = tanh-GELU, 2 = +residual.
template<int EPI, bool BF16W>
__global__ __launch_bounds__(256)
void gemm_kernel(const float* __restrict__ A, const void* __restrict__ Wv,
                 const float* __restrict__ bias, const float* __restrict__ resid,
                 float* __restrict__ out, int N, int K) {
  __shared__ __align__(16) bf16 As[2][BM * LDT];
  __shared__ __align__(16) bf16 Bs[2][BN * LDT];
  const int tid  = threadIdx.x;
  const int lane = tid & 31;
  const int wv   = tid >> 5;
  const int hgrp = lane >> 4;       // half-wave: 0 or 1
  const int l16  = lane & 15;
  const int m0 = blockIdx.y * BM;
  const int n0 = blockIdx.x * BN;
  const int waveM = (wv & 3) * 16;  // 4 wave-rows
  const int waveN = (wv >> 2) * 32; // 2 wave-cols, 2 accumulators each

  const int srow = tid >> 2;        // 0..63
  const int scol = (tid & 3) * 8;   // 0,8,16,24

  const bf16*  Wb = (const bf16*)Wv;
  const float* Wf = (const float*)Wv;

  f32x8 acc0 = {0.f,0.f,0.f,0.f,0.f,0.f,0.f,0.f};
  f32x8 acc1 = {0.f,0.f,0.f,0.f,0.f,0.f,0.f,0.f};

  const int nt = K / BK;

#if HAVE_TDM
  if constexpr (BF16W) {
    if (wv == 0) tdm_load_tile(Wb + (size_t)n0 * K, &Bs[0][0], K);  // prologue: tile 0
  }
#endif

  for (int it = 0; it < nt; ++it) {
    const int k0 = it * BK;
    bf16* as = &As[it & 1][0];
    bf16* bs = &Bs[it & 1][0];

    // ---- stage A tile fp32 -> bf16 (8 contiguous elems per thread) ----
    {
      const float* ap = A + (size_t)(m0 + srow) * K + (k0 + scol);
      bf16* dst = as + srow * LDT + scol;
      #pragma unroll
      for (int i = 0; i < 8; ++i) dst[i] = (bf16)ap[i];
      if (k0 + BK < K) __builtin_prefetch(ap + BK, 0, 0);   // global_prefetch_b8
    }

    // ---- stage B tile ----
    if constexpr (!BF16W) {
      const int n = n0 + srow;
      bf16* dst = bs + srow * LDT + scol;
      if (n < N) {
        const float* wp = Wf + (size_t)n * K + (k0 + scol);
        #pragma unroll
        for (int i = 0; i < 8; ++i) dst[i] = (bf16)wp[i];
        if (k0 + BK < K) __builtin_prefetch(wp + BK, 0, 0);
      } else {
        #pragma unroll
        for (int i = 0; i < 8; ++i) dst[i] = (bf16)0.0f;
      }
    }
#if !HAVE_TDM
    if constexpr (BF16W) {  // fallback copy if TDM builtin unavailable
      const bf16x8* wp = (const bf16x8*)(Wb + (size_t)(n0 + srow) * K + k0) + (tid & 3);
      *(bf16x8*)(bs + srow * LDT + scol) = *wp;
    }
#else
    if constexpr (BF16W) {
      if (wv == 0) wait_tensor0();  // tile `it` has landed in bs
    }
#endif
    __syncthreads();

#if HAVE_TDM
    if constexpr (BF16W) {
      // overlap next tile's DMA with this tile's WMMA (safe: all waves passed barrier,
      // so reads of Bs[(it+1)&1] from step it-1 are complete)
      if (wv == 0 && it + 1 < nt)
        tdm_load_tile(Wb + (size_t)n0 * K + (size_t)(it + 1) * BK, &Bs[(it + 1) & 1][0], K);
    }
#endif

    // A fragment (16x32 bf16: M = lane%16, K sequential per half-wave, 0-15 vs 16-31)
    const bf16* ap = as + (waveM + l16) * LDT + hgrp * 16;
    bf16x16 afrag = SHUF16(*(const bf16x8*)(ap), *(const bf16x8*)(ap + 8));

    // B fragments (32x16 bf16: N = lane%16, K sequential per half-wave)
    const bf16* b0 = bs + (waveN + l16) * LDT + hgrp * 16;
    bf16x16 bf0 = SHUF16(*(const bf16x8*)(b0), *(const bf16x8*)(b0 + 8));
    const bf16* b1 = bs + (waveN + 16 + l16) * LDT + hgrp * 16;
    bf16x16 bf1 = SHUF16(*(const bf16x8*)(b1), *(const bf16x8*)(b1 + 8));

    acc0 = wmma_bf16(afrag, bf0, acc0);
    acc1 = wmma_bf16(afrag, bf1, acc1);
    // no trailing barrier: A/B are double-buffered (parity resolves WAR)
  }

  // epilogue: C/D layout row = j + 8*hgrp, col = lane%16
  #pragma unroll
  for (int j = 0; j < 8; ++j) {
    const int r  = m0 + waveM + j + hgrp * 8;
    const int c0 = n0 + waveN + l16;
    const int c1 = c0 + 16;
    float v0 = acc0[j], v1 = acc1[j];
    if (bias) {
      if (c0 < N) v0 += bias[c0];
      if (c1 < N) v1 += bias[c1];
    }
    if (EPI == 1) { v0 = gelu_tanh(v0); v1 = gelu_tanh(v1); }
    if (EPI == 2) {
      if (c0 < N) v0 += resid[(size_t)r * N + c0];
      if (c1 < N) v1 += resid[(size_t)r * N + c1];
    }
    if (c0 < N) out[(size_t)r * N + c0] = v0;
    if (c1 < N) out[(size_t)r * N + c1] = v1;
  }
}

// ---------------- flash attention: one wave per 16-query tile per (b,h) ----------------
constexpr int PLD = 40; // 32 keys + pad, halves

__global__ __launch_bounds__(256)
void attn_kernel(const float* __restrict__ qkv, float* __restrict__ y) {
  __shared__ __align__(16) bf16 Pbuf[8][16 * PLD];
  const int tid = threadIdx.x, lane = tid & 31, wv = tid >> 5;
  const int gw = blockIdx.x * 8 + wv;
  const int QT = Tz / 16;                       // 64 query tiles
  const int qt = gw % QT;
  const int h  = (gw / QT) % Hz;
  const int b  = gw / (QT * Hz);
  const int q0 = qt * 16;
  const int hgrp = lane >> 4, l16 = lane & 15;
  const size_t rs = (size_t)(3 * Cz);           // qkv row stride
  const float* qbase = qkv + (size_t)b * Tz * rs + (size_t)h * HDz;
  const float* kbase = qbase + Cz;
  const float* vbase = qbase + 2 * Cz;
  const float scale = 0.125f;                   // 1/sqrt(64), folded into Q

  // resident Q fragments: 16x64 = two 16x32 A-fragments
  bf16x16 qf[2];
  {
    const float* qr = qbase + (size_t)(q0 + l16) * rs;
    #pragma unroll
    for (int kk = 0; kk < 2; ++kk) {
      const int base = kk * 32 + hgrp * 16;
      bf16x8 lo, hi;
      #pragma unroll
      for (int i = 0; i < 8; ++i) {
        lo[i] = (bf16)(qr[base + i] * scale);
        hi[i] = (bf16)(qr[base + 8 + i] * scale);
      }
      qf[kk] = SHUF16(lo, hi);
    }
  }

  const f32x8 vzero = {0.f,0.f,0.f,0.f,0.f,0.f,0.f,0.f};
  f32x8 o0 = vzero, o1 = vzero, o2 = vzero, o3 = vzero;   // O: 16x64
  float mrow[8], lrow[8];
  #pragma unroll
  for (int j = 0; j < 8; ++j) { mrow[j] = -1e30f; lrow[j] = 0.f; }

  bf16* P = &Pbuf[wv][0];
  const int kend = q0 + 16;                     // causal: keys < q0+16
  for (int kt = 0; kt < kend; kt += 32) {
    f32x8 s0 = vzero, s1 = vzero;               // S: 16 q x 32 keys
    #pragma unroll
    for (int sub = 0; sub < 2; ++sub) {
      const float* kr = kbase + (size_t)(kt + sub * 16 + l16) * rs;  // key row
      #pragma unroll
      for (int kk = 0; kk < 2; ++kk) {
        const int base = kk * 32 + hgrp * 16;
        bf16x8 lo, hi;
        #pragma unroll
        for (int i = 0; i < 8; ++i) { lo[i] = (bf16)kr[base + i]; hi[i] = (bf16)kr[base + 8 + i]; }
        bf16x16 kf = SHUF16(lo, hi);
        if (sub == 0) s0 = wmma_bf16(qf[kk], kf, s0);
        else          s1 = wmma_bf16(qf[kk], kf, s1);
      }
    }
    // online softmax (row reductions across 16-lane half-wave groups)
    const bool needmask = (kt + 31) > q0;
    #pragma unroll
    for (int j = 0; j < 8; ++j) {
      const int qrow = q0 + j + hgrp * 8;
      float v0 = s0[j], v1 = s1[j];
      if (needmask) {
        if (kt + l16      > qrow) v0 = -1e30f;
        if (kt + 16 + l16 > qrow) v1 = -1e30f;
      }
      float mx = fmaxf(v0, v1);
      #pragma unroll
      for (int off = 8; off >= 1; off >>= 1) mx = fmaxf(mx, __shfl_xor(mx, off, 32));
      const float mnew  = fmaxf(mrow[j], mx);
      const float alpha = __expf(mrow[j] - mnew);
      const float p0 = __expf(v0 - mnew);
      const float p1 = __expf(v1 - mnew);
      float rsum = p0 + p1;
      #pragma unroll
      for (int off = 8; off >= 1; off >>= 1) rsum += __shfl_xor(rsum, off, 32);
      lrow[j] = lrow[j] * alpha + rsum;
      mrow[j] = mnew;
      o0[j] *= alpha; o1[j] *= alpha; o2[j] *= alpha; o3[j] *= alpha;
      const int prow = j + hgrp * 8;
      P[prow * PLD + l16]      = (bf16)p0;
      P[prow * PLD + 16 + l16] = (bf16)p1;
    }
    // wave-local LDS RAW: wait on DScnt before reading P back in A-fragment order
    asm volatile("s_wait_dscnt 0" ::: "memory");
    const bf16* pr = &P[l16 * PLD + hgrp * 16];
    bf16x16 pf = SHUF16(*(const bf16x8*)(pr), *(const bf16x8*)(pr + 8));
    // O += P(16x32) @ V(32x64): 4 hd-subtiles
    #pragma unroll
    for (int sub = 0; sub < 4; ++sub) {
      const float* vc = vbase + sub * 16 + l16;
      bf16x8 lo, hi;
      #pragma unroll
      for (int i = 0; i < 8; ++i) {
        lo[i] = (bf16)vc[(size_t)(kt + hgrp * 16 + i)     * rs];
        hi[i] = (bf16)vc[(size_t)(kt + hgrp * 16 + 8 + i) * rs];
      }
      bf16x16 vf = SHUF16(lo, hi);
      if      (sub == 0) o0 = wmma_bf16(pf, vf, o0);
      else if (sub == 1) o1 = wmma_bf16(pf, vf, o1);
      else if (sub == 2) o2 = wmma_bf16(pf, vf, o2);
      else               o3 = wmma_bf16(pf, vf, o3);
    }
  }
  // normalize and write y[b, q0+row, h*64 + col]
  float* yb = y + ((size_t)b * Tz + q0) * Cz + (size_t)h * HDz;
  #pragma unroll
  for (int j = 0; j < 8; ++j) {
    const int row = j + hgrp * 8;
    const float inv = 1.0f / lrow[j];
    float* orow = yb + (size_t)row * Cz;
    orow[l16]      = o0[j] * inv;
    orow[16 + l16] = o1[j] * inv;
    orow[32 + l16] = o2[j] * inv;
    orow[48 + l16] = o3[j] * inv;
  }
}

// ---------------- host-side orchestration ----------------
extern "C" void kernel_launch(void* const* d_in, const int* in_sizes, int n_in,
                              void* d_out, int out_size, void* d_ws, size_t ws_size,
                              hipStream_t stream) {
  (void)in_sizes; (void)n_in; (void)out_size; (void)ws_size;
  const int*   idx    = (const int*)  d_in[0];
  const float* wte    = (const float*)d_in[1];
  const float* wpe    = (const float*)d_in[2];
  const float* ln1_s  = (const float*)d_in[3];
  const float* ln1_b  = (const float*)d_in[4];
  const float* attn_w = (const float*)d_in[5];
  const float* attn_b = (const float*)d_in[6];
  const float* proj_w = (const float*)d_in[7];
  const float* proj_b = (const float*)d_in[8];
  const float* ln2_s  = (const float*)d_in[9];
  const float* ln2_b  = (const float*)d_in[10];
  const float* fc_w   = (const float*)d_in[11];
  const float* fc_b   = (const float*)d_in[12];
  const float* fc2_w  = (const float*)d_in[13];
  const float* fc2_b  = (const float*)d_in[14];
  const float* lnf_s  = (const float*)d_in[15];
  const float* lnf_b  = (const float*)d_in[16];

  // workspace layout (fp32): x | ln-out | qkv | attn-out | fc-out | bf16 weight slot
  float* x    = (float*)d_ws;
  float* hbuf = x    + (size_t)Mz * Cz;
  float* qkv  = hbuf + (size_t)Mz * Cz;
  float* yatt = qkv  + (size_t)Mz * 3 * Cz;
  float* fcb  = yatt + (size_t)Mz * Cz;
  bf16*  wbt  = (bf16*)(fcb + (size_t)Mz * 4 * Cz);   // up to C*4C bf16 (~4.7 MB)

  const dim3 blk(256);
  auto convblocks = [](int K, int N) { return (N * (K / 8) + 255) / 256; };

  embed_kernel<<<Mz, blk, 0, stream>>>(idx, wte, wpe, x);

  for (int l = 0; l < Lz; ++l) {
    const float* aw  = attn_w + (size_t)l * Cz * 3 * Cz;
    const float* ab  = attn_b + (size_t)l * 3 * Cz;
    const float* pw  = proj_w + (size_t)l * Cz * Cz;
    const float* pb  = proj_b + (size_t)l * Cz;
    const float* fw  = fc_w   + (size_t)l * Cz * 4 * Cz;
    const float* fb  = fc_b   + (size_t)l * 4 * Cz;
    const float* f2w = fc2_w  + (size_t)l * 4 * Cz * Cz;
    const float* f2b = fc2_b  + (size_t)l * Cz;

    ln_kernel<<<Mz, blk, 0, stream>>>(x, ln1_s + (size_t)l * Cz, ln1_b + (size_t)l * Cz, hbuf, 1e-5f);
    wconv_kernel<<<convblocks(Cz, 3 * Cz), blk, 0, stream>>>(aw, wbt, Cz, 3 * Cz);
    gemm_kernel<0,true><<<dim3((3 * Cz) / BN, Mz / BM), blk, 0, stream>>>(hbuf, wbt, ab, nullptr, qkv, 3 * Cz, Cz);
    attn_kernel<<<(Bz * Hz * (Tz / 16)) / 8, blk, 0, stream>>>(qkv, yatt);
    wconv_kernel<<<convblocks(Cz, Cz), blk, 0, stream>>>(pw, wbt, Cz, Cz);
    gemm_kernel<2,true><<<dim3(Cz / BN, Mz / BM), blk, 0, stream>>>(yatt, wbt, pb, x, x, Cz, Cz);
    ln_kernel<<<Mz, blk, 0, stream>>>(x, ln2_s + (size_t)l * Cz, ln2_b + (size_t)l * Cz, hbuf, 1e-5f);
    wconv_kernel<<<convblocks(Cz, 4 * Cz), blk, 0, stream>>>(fw, wbt, Cz, 4 * Cz);
    gemm_kernel<1,true><<<dim3((4 * Cz) / BN, Mz / BM), blk, 0, stream>>>(hbuf, wbt, fb, nullptr, fcb, 4 * Cz, Cz);
    wconv_kernel<<<convblocks(4 * Cz, Cz), blk, 0, stream>>>(f2w, wbt, 4 * Cz, Cz);
    gemm_kernel<2,true><<<dim3(Cz / BN, Mz / BM), blk, 0, stream>>>(fcb, wbt, f2b, x, x, Cz, 4 * Cz);
  }

  ln_kernel<<<Mz, blk, 0, stream>>>(x, lnf_s, lnf_b, hbuf, 1e-6f);
  // logits = h @ wte^T : wte already [N=V, K=C], fp32 on-the-fly path
  gemm_kernel<0,false><<<dim3((Vz + BN - 1) / BN, Mz / BM), blk, 0, stream>>>(
      hbuf, wte, nullptr, nullptr, (float*)d_out, Vz, Cz);
}
